// LstmModelConv_29686813950487
// MI455X (gfx1250) — matrix operand order repaired and checked by reading, not compiled, not run on hardware
//
#include <hip/hip_runtime.h>

// ---------------- types ----------------
using bf16 = __bf16;
typedef bf16  v16bf __attribute__((ext_vector_type(16)));
typedef bf16  v8bf  __attribute__((ext_vector_type(8)));
typedef float v8f   __attribute__((ext_vector_type(8)));
typedef float v4f   __attribute__((ext_vector_type(4)));

// ---------------- problem constants ----------------
#define NB   1024       // batch
#define NT_  512        // seq len
#define NS   4          // sensors
#define NH   640        // hidden
#define G4   2560       // 4*H
#define BT   64         // batch tile per workgroup
#define PADW 656        // padded LDS row stride (bf16 elems)
#define HELEMS (BT * PADW)
#define KT   20         // K tiles (640/32)
#define NTILE 160       // gate-column tiles per sensor (2560/16)

// scratch partition (bytes)
#define WP_BYTES   13107200ULL                 // 4*160*20*32*16 bf16
#define BWS_OFF    WP_BYTES                    // 10240 floats
#define CWS_OFF    (WP_BYTES + 40960ULL)       // 64*160*256 floats
#define EMB_OFF    (CWS_OFF + 10485760ULL)     // 1024*4*3 floats

// ---------------- fast transcendentals (prefer HW v_tanh_f32 on gfx1250) ----
#if __has_builtin(__builtin_amdgcn_tanhf)
__device__ __forceinline__ float fast_tanh(float v) {
    return __builtin_amdgcn_tanhf(v);
}
#elif __has_builtin(__builtin_amdgcn_tanh_f32)
__device__ __forceinline__ float fast_tanh(float v) {
    return __builtin_amdgcn_tanh_f32(v);
}
#else
__device__ __forceinline__ float fast_tanh(float v) {
    return 1.0f - 2.0f * __builtin_amdgcn_rcpf(1.0f + __expf(2.0f * v));
}
#endif
__device__ __forceinline__ float fast_sig(float v) {
    // sigmoid(x) = 0.5*tanh(x/2) + 0.5 : one TRANS op + one FMA
    return fast_tanh(0.5f * v) * 0.5f + 0.5f;
}

// ---------------- pack W_hh (fp32 row-major) -> bf16 B-fragment-major ----------------
// Fragment (sensor, ntile, ktile): lane l<16 holds W[nt*16 + l][kt*32 + 0..15]
//                                  lane l>=16 holds W[nt*16 + (l-16)][kt*32 + 16..31]
__global__ __launch_bounds__(256) void pack_w_kernel(const float* __restrict__ whh,
                                                     bf16* __restrict__ wp) {
    int tid = blockIdx.x * blockDim.x + threadIdx.x;       // 409600
    if (tid >= NS * NTILE * KT * 32) return;
    int l  = tid & 31;
    int kt = (tid >> 5) % KT;
    int nt = (tid / (32 * KT)) % NTILE;
    int s  = tid / (32 * KT * NTILE);
    int g  = nt * 16 + (l & 15);
    int k0 = kt * 32 + (l >> 4) * 16;
    const float* src = whh + ((size_t)(s * G4 + g)) * NH + k0;
    v16bf v;
#pragma unroll
    for (int i = 0; i < 16; ++i) v[i] = (bf16)src[i];
    *(v16bf*)(wp + (size_t)tid * 16) = v;
}

__global__ __launch_bounds__(256) void pack_bias_kernel(const float* __restrict__ bih,
                                                        const float* __restrict__ bhh,
                                                        float* __restrict__ bws) {
    int i = blockIdx.x * blockDim.x + threadIdx.x;
    if (i < NS * G4) bws[i] = bih[i] + bhh[i];
}

// ---------------- main LSTM kernel ----------------
// One WG = (sensor, 64 batch rows), full T recurrence in-kernel.
// Each wave owns ALL 4 M-tiles and 5 gate-column tiles -> every weight
// fragment is loaded exactly once per WG per step (no redundancy).
__global__ __launch_bounds__(256) void lstm_wmma_kernel(
    const float* __restrict__ x,      // (B, T, S)
    const float* __restrict__ win,    // (S, 4H)
    const bf16*  __restrict__ wp,     // packed weights
    const float* __restrict__ bws,    // (S, 4H) combined bias
    float*       __restrict__ cws,    // cell-state scratch
    float*       __restrict__ embws,  // (B, S, 3)
    const float* __restrict__ fcw,    // (3, H)
    const float* __restrict__ fcb)    // (3,)
{
    extern __shared__ char smem[];
    bf16*  hb = (bf16*)smem;                                // [2][BT][PADW]
    float* xs = (float*)(smem + 2 * HELEMS * sizeof(bf16)); // [2][BT]

    const int tid  = threadIdx.x;
    const int lane = tid & 31;
    const int jq   = tid >> 5;        // wave id = which 5 gate-column tiles
    const int col  = lane & 15;
    const int hs   = lane >> 4;
    const int moff = hs * 8;

    const int sensor = blockIdx.x & 3;
    const int b0     = (blockIdx.x >> 2) * BT;

    for (int i = tid; i < HELEMS; i += 256) hb[i] = (bf16)0.0f;
    if (tid < BT) xs[tid] = x[(size_t)(b0 + tid) * (NT_ * NS) + sensor];
    __syncthreads();

    const v16bf* wpv = (const v16bf*)wp + (size_t)sensor * (NTILE * KT * 32);

    for (int t = 0; t < NT_; ++t) {
        const int cur = t & 1, nxt = cur ^ 1;
        const bf16* hcur = hb + cur * HELEMS;
        bf16*       hnxt = hb + nxt * HELEMS;
        const float* xcur = xs + cur * BT;

        // A-fragment base pointers for the 4 M tiles (this thread's lane view)
        const bf16* hrm[4];
#pragma unroll
        for (int m = 0; m < 4; ++m)
            hrm[m] = hcur + (m * 16 + col) * PADW + hs * 16;

#pragma unroll 1
        for (int jl = 0; jl < 5; ++jl) {
            const int j = jq * 5 + jl;

            float wv[4], bv[4];
#pragma unroll
            for (int gate = 0; gate < 4; ++gate) {
                const int g = gate * NH + j * 16 + col;
                wv[gate] = win[sensor * G4 + g];
                bv[gate] = bws[sensor * G4 + g];
            }

            // init accumulators with x_t * w_in + bias (C-fragment layout)
            v8f acc[4][4];
#pragma unroll
            for (int m = 0; m < 4; ++m) {
                const int rbase = m * 16 + moff;
#pragma unroll
                for (int r = 0; r < 8; ++r) {
                    const float xr = xcur[rbase + r];
#pragma unroll
                    for (int gate = 0; gate < 4; ++gate)
                        acc[m][gate][r] = xr * wv[gate] + bv[gate];
                }
            }

            // hoisted per-gate B base pointers; kt advances by constant stride
            const v16bf* bp[4];
#pragma unroll
            for (int gate = 0; gate < 4; ++gate)
                bp[gate] = wpv + ((size_t)(gate * 40 + j) * KT) * 32 + lane;

            // K reduction: gates += h_{t-1} * W_hh^T   (16 WMMAs / kt)
#pragma unroll 4
            for (int kt = 0; kt < KT; ++kt) {
                v16bf a[4];
#pragma unroll
                for (int m = 0; m < 4; ++m) {
                    v8bf lo = *(const v8bf*)(hrm[m] + kt * 32);
                    v8bf hi = *(const v8bf*)(hrm[m] + kt * 32 + 8);
                    a[m] = __builtin_shufflevector(lo, hi,
                        0,1,2,3,4,5,6,7,8,9,10,11,12,13,14,15);
                }
#pragma unroll
                for (int gate = 0; gate < 4; ++gate) {
                    v16bf bfr = bp[gate][kt * 32];
#pragma unroll
                    for (int m = 0; m < 4; ++m)
                        acc[m][gate] = __builtin_amdgcn_wmma_f32_16x16x32_bf16(
                            false, a[m], false, bfr, (short)0, acc[m][gate],
                            false, false);
                }
            }

            // pointwise LSTM cell update
#pragma unroll
            for (int m = 0; m < 4; ++m) {
                float* cp = cws + ((size_t)blockIdx.x * 160 + (size_t)(m * 40 + j)) * 256
                                + lane * 8;
                v4f c_lo = {0.f, 0.f, 0.f, 0.f}, c_hi = {0.f, 0.f, 0.f, 0.f};
                if (t > 0) { c_lo = *(const v4f*)cp; c_hi = *(const v4f*)(cp + 4); }
                float cn[8];
#pragma unroll
                for (int r = 0; r < 8; ++r) {
                    const float cprev = (r < 4) ? c_lo[r & 3] : c_hi[r & 3];
                    const float iv = fast_sig(acc[m][0][r]);
                    const float fv = fast_sig(acc[m][1][r]);
                    const float gv = fast_tanh(acc[m][2][r]);
                    const float ov = fast_sig(acc[m][3][r]);
                    const float c2 = fv * cprev + iv * gv;
                    cn[r] = c2;
                    const float hv = ov * fast_tanh(c2);
                    hnxt[(m * 16 + moff + r) * PADW + j * 16 + col] = (bf16)hv;
                }
                v4f s_lo = {cn[0], cn[1], cn[2], cn[3]};
                v4f s_hi = {cn[4], cn[5], cn[6], cn[7]};
                *(v4f*)cp = s_lo;
                *(v4f*)(cp + 4) = s_hi;
            }
        }

        // stage x for next step into the other slot
        if (t + 1 < NT_ && tid < BT)
            xs[(nxt)*BT + tid] =
                x[(size_t)(b0 + tid) * (NT_ * NS) + (size_t)(t + 1) * NS + sensor];
        __syncthreads();
    }

    // FC head off final h (T even -> final h is buffer 0)
    const bf16* hf = hb;
    if (tid < BT * 3) {
        const int row = tid / 3, o = tid - row * 3;
        float a = fcb[o];
        for (int hh = 0; hh < NH; ++hh)
            a += (float)hf[row * PADW + hh] * fcw[o * NH + hh];
        embws[((size_t)(b0 + row) * NS + sensor) * 3 + o] = a;
    }
}

// ---------------- folded conv + mean ----------------
__global__ __launch_bounds__(256) void finalize_kernel(const float* __restrict__ embws,
                                                       const float* __restrict__ convw,
                                                       const float* __restrict__ convb,
                                                       float* __restrict__ out) {
    int b = blockIdx.x * blockDim.x + threadIdx.x;
    if (b >= NB) return;
    float wsum[9];
#pragma unroll
    for (int k = 0; k < 9; ++k) {
        float s = 0.f;
        for (int co = 0; co < 32; ++co) s += convw[co * 9 + k];
        wsum[k] = s;
    }
    float bsum = 0.f;
    for (int co = 0; co < 32; ++co) bsum += convb[co];

    float feat[3][6];
#pragma unroll
    for (int kh = 0; kh < 3; ++kh) {
#pragma unroll
        for (int c = 0; c < 4; ++c)
            feat[kh][c] = embws[((size_t)b * NS + c) * 3 + kh];
        feat[kh][4] = 1.0f;
        feat[kh][5] = 1.0f;
    }
#pragma unroll
    for (int w = 0; w < 4; ++w) {
        float a = bsum;
#pragma unroll
        for (int kh = 0; kh < 3; ++kh)
#pragma unroll
            for (int kw = 0; kw < 3; ++kw)
                a += feat[kh][w + kw] * wsum[kh * 3 + kw];
        out[b * 4 + w] = a * (1.0f / 32.0f);
    }
}

// ---------------- launcher ----------------
extern "C" void kernel_launch(void* const* d_in, const int* in_sizes, int n_in,
                              void* d_out, int out_size, void* d_ws, size_t ws_size,
                              hipStream_t stream) {
    const float* x     = (const float*)d_in[0];
    const float* wih   = (const float*)d_in[1];
    const float* whh   = (const float*)d_in[2];
    const float* bih   = (const float*)d_in[3];
    const float* bhh   = (const float*)d_in[4];
    const float* fcw   = (const float*)d_in[5];
    const float* fcb   = (const float*)d_in[6];
    const float* convw = (const float*)d_in[7];
    const float* convb = (const float*)d_in[8];
    float* out = (float*)d_out;

    bf16*  wpk   = (bf16*)d_ws;
    float* bws   = (float*)((char*)d_ws + BWS_OFF);
    float* cws   = (float*)((char*)d_ws + CWS_OFF);
    float* embws = (float*)((char*)d_ws + EMB_OFF);

    pack_w_kernel<<<(NS * NTILE * KT * 32 + 255) / 256, 256, 0, stream>>>(whh, wpk);
    pack_bias_kernel<<<(NS * G4 + 255) / 256, 256, 0, stream>>>(bih, bhh, bws);

    const size_t smem = 2ULL * HELEMS * sizeof(bf16) + 2ULL * BT * sizeof(float);
    lstm_wmma_kernel<<<(NB / BT) * NS, 256, smem, stream>>>(
        x, wih, wpk, bws, cws, embws, fcw, fcb);

    finalize_kernel<<<(NB + 255) / 256, 256, 0, stream>>>(embws, convw, convb, out);
}
